// AGCNBlock_5360119186049
// MI455X (gfx1250) — matrix operand order repaired and checked by reading, not compiled, not run on hardware
//
#include <hip/hip_runtime.h>
#include <hip/hip_bf16.h>
#include <math.h>

// Problem constants (match reference)
#define BB   8
#define NN   2048
#define DH   64
#define KMAX 512
#define EPSV 1e-10f

typedef __attribute__((ext_vector_type(16))) __bf16 v16bf;
typedef __attribute__((ext_vector_type(8)))  float  v8f;

__device__ __forceinline__ __bf16 f2bf(float x) {
  union { float f; unsigned u; } c; c.f = x;
  unsigned r = c.u + 0x7fffu + ((c.u >> 16) & 1u);   // round-to-nearest-even
  unsigned short h = (unsigned short)(r >> 16);
  union { unsigned short s; __bf16 b; } o; o.s = h; return o.b;
}

// ---------------- elementwise f32 -> bf16 ----------------
__global__ void k_cvt_bf16(const float* __restrict__ src, __bf16* __restrict__ dst, long n) {
  long i = (long)blockIdx.x * blockDim.x + threadIdx.x;
  if (i < n) dst[i] = f2bf(src[i]);
}

// ---------------- transposing f32 -> bf16 (32x32 LDS tiles, coalesced) ----------------
// src: (R x Cc) row-major per batch; dst: (Cc x R) row-major per batch.
__global__ void k_cvt_bf16_t(const float* __restrict__ src, __bf16* __restrict__ dst,
                             int R, int Cc, long sS, long sD) {
  __shared__ __bf16 tile[32][33];
  int b  = blockIdx.z;
  int r0 = blockIdx.y << 5, c0 = blockIdx.x << 5;
  int tx = threadIdx.x & 31, ty = threadIdx.x >> 5;    // 256 threads = 32 x 8
  const float* S = src + (long)b * sS;
  __bf16*      D = dst + (long)b * sD;
#pragma unroll
  for (int i = ty; i < 32; i += 8)
    tile[i][tx] = f2bf(S[(long)(r0 + i) * Cc + (c0 + tx)]);
  __syncthreads();
#pragma unroll
  for (int i = ty; i < 32; i += 8)
    D[(long)(c0 + i) * R + (r0 + tx)] = tile[tx][i];
}

// ---------------- WMMA fragment loads (contiguous) ----------------
__device__ __forceinline__ void loadA(v16bf& a, const __bf16* __restrict__ p) {
#pragma unroll
  for (int j = 0; j < 16; ++j) a[j] = p[j];
}
__device__ __forceinline__ void loadB(v16bf& f, const __bf16* __restrict__ p) {
#pragma unroll
  for (int j = 0; j < 16; ++j) f[j] = p[j];
}

// ---------------- batched WMMA GEMM: C = A(MxK) * B(KxN) (+bias, opt tanh) ----------
// B supplied TRANSPOSED: Bt (N x K) row-major -> all fragment loads contiguous.
// Block = 128 threads (4 waves); block computes a 64(M) x 64(N) tile; wave w owns
// rows [m0+16w, m0+16w+16) x 4 column tiles. K must be a multiple of 64:
// two-deep software pipeline (register double buffer) hides load latency.
__global__ void __launch_bounds__(128, 1)
k_wmma_gemm(const __bf16* __restrict__ A, const __bf16* __restrict__ Bt,
            float* __restrict__ C, const float* __restrict__ bias,
            int K, int lda, int ldbt, int ldc,
            long sA, long sB, long sC, int doTanh)
{
  int wid  = threadIdx.x >> 5;
  int lane = threadIdx.x & 31;
  int row  = lane & 15, half = lane >> 4;

  int b      = blockIdx.z;
  int m0     = (blockIdx.y << 6) + (wid << 4);
  int n0base = blockIdx.x << 6;

  const __bf16* ap  = A  + (long)b * sA + (long)(m0 + row) * lda + (half << 4);
  const __bf16* Bb  = Bt + (long)b * sB;
  const __bf16* bp0 = Bb + (long)(n0base +  0 + row) * ldbt + (half << 4);
  const __bf16* bp1 = Bb + (long)(n0base + 16 + row) * ldbt + (half << 4);
  const __bf16* bp2 = Bb + (long)(n0base + 32 + row) * ldbt + (half << 4);
  const __bf16* bp3 = Bb + (long)(n0base + 48 + row) * ldbt + (half << 4);

  v8f acc0 = {}, acc1 = {}, acc2 = {}, acc3 = {};
  v16bf a0, b00, b01, b02, b03;      // pipeline set 0 (even k-chunk)
  v16bf a1, b10, b11, b12, b13;      // pipeline set 1 (odd  k-chunk)

  // prologue: k = 0
  loadA(a0, ap);
  loadB(b00, bp0); loadB(b01, bp1); loadB(b02, bp2); loadB(b03, bp3);
  __builtin_prefetch((const void*)(ap + 512), 0, 1);

  int nIt = K >> 6;                  // K % 64 == 0 for all call sites
  for (int it = 0; it < nIt - 1; ++it) {
    // issue set-1 loads (k+32) before consuming set-0 -> partial loadcnt waits
    loadA(a1, ap + 32);
    loadB(b10, bp0 + 32); loadB(b11, bp1 + 32); loadB(b12, bp2 + 32); loadB(b13, bp3 + 32);
    acc0 = __builtin_amdgcn_wmma_f32_16x16x32_bf16(false, a0, false, b00, (short)0, acc0, false, false);
    acc1 = __builtin_amdgcn_wmma_f32_16x16x32_bf16(false, a0, false, b01, (short)0, acc1, false, false);
    acc2 = __builtin_amdgcn_wmma_f32_16x16x32_bf16(false, a0, false, b02, (short)0, acc2, false, false);
    acc3 = __builtin_amdgcn_wmma_f32_16x16x32_bf16(false, a0, false, b03, (short)0, acc3, false, false);

    ap += 64; bp0 += 64; bp1 += 64; bp2 += 64; bp3 += 64;

    loadA(a0, ap);
    loadB(b00, bp0); loadB(b01, bp1); loadB(b02, bp2); loadB(b03, bp3);
    __builtin_prefetch((const void*)(ap + 512), 0, 1);
    acc0 = __builtin_amdgcn_wmma_f32_16x16x32_bf16(false, a1, false, b10, (short)0, acc0, false, false);
    acc1 = __builtin_amdgcn_wmma_f32_16x16x32_bf16(false, a1, false, b11, (short)0, acc1, false, false);
    acc2 = __builtin_amdgcn_wmma_f32_16x16x32_bf16(false, a1, false, b12, (short)0, acc2, false, false);
    acc3 = __builtin_amdgcn_wmma_f32_16x16x32_bf16(false, a1, false, b13, (short)0, acc3, false, false);
  }
  // epilogue: last 64-K chunk
  loadA(a1, ap + 32);
  loadB(b10, bp0 + 32); loadB(b11, bp1 + 32); loadB(b12, bp2 + 32); loadB(b13, bp3 + 32);
  acc0 = __builtin_amdgcn_wmma_f32_16x16x32_bf16(false, a0, false, b00, (short)0, acc0, false, false);
  acc1 = __builtin_amdgcn_wmma_f32_16x16x32_bf16(false, a0, false, b01, (short)0, acc1, false, false);
  acc2 = __builtin_amdgcn_wmma_f32_16x16x32_bf16(false, a0, false, b02, (short)0, acc2, false, false);
  acc3 = __builtin_amdgcn_wmma_f32_16x16x32_bf16(false, a0, false, b03, (short)0, acc3, false, false);
  acc0 = __builtin_amdgcn_wmma_f32_16x16x32_bf16(false, a1, false, b10, (short)0, acc0, false, false);
  acc1 = __builtin_amdgcn_wmma_f32_16x16x32_bf16(false, a1, false, b11, (short)0, acc1, false, false);
  acc2 = __builtin_amdgcn_wmma_f32_16x16x32_bf16(false, a1, false, b12, (short)0, acc2, false, false);
  acc3 = __builtin_amdgcn_wmma_f32_16x16x32_bf16(false, a1, false, b13, (short)0, acc3, false, false);

  float* cb = C + (long)b * sC + (long)(m0 + (half << 3)) * ldc + n0base + row;
#pragma unroll
  for (int nt = 0; nt < 4; ++nt) {
    const v8f& acc = nt == 0 ? acc0 : nt == 1 ? acc1 : nt == 2 ? acc2 : acc3;
    int nc = (nt << 4);
    float bv = bias ? bias[n0base + nc + row] : 0.0f;
    float* cp = cb + nc;
#pragma unroll
    for (int v = 0; v < 8; ++v) {
      float val = acc[v] + bv;
      if (doTanh) val = tanhf(val);
      cp[(long)v * ldc] = val;
    }
  }
}

// ---------------- small kernels ----------------
__global__ void k_apply_mask(float* __restrict__ h, const float* __restrict__ mask) {
  long i = (long)blockIdx.x * blockDim.x + threadIdx.x;
  if (i < (long)BB * NN * DH) h[i] *= mask[i / DH];
}

__global__ void k_masksum(const float* __restrict__ mask, float* __restrict__ msum) {
  __shared__ float red[256];
  int b = blockIdx.x;
  float s = 0.f;
  for (int n = threadIdx.x; n < NN; n += blockDim.x) s += mask[b * NN + n];
  red[threadIdx.x] = s; __syncthreads();
  for (int o = 128; o; o >>= 1) { if (threadIdx.x < o) red[threadIdx.x] += red[threadIdx.x + o]; __syncthreads(); }
  if (threadIdx.x == 0) msum[b] = red[0];
}

__global__ void k_colmean(const float* __restrict__ h, const float* __restrict__ msum,
                          float* __restrict__ out) {
  int b = blockIdx.x, d = threadIdx.x;
  float s = 0.f;
  for (int n = 0; n < NN; ++n) s += h[((long)b * NN + n) * DH + d];
  out[b * DH + d] = s / (EPSV + msum[b]);
}

__global__ void k_att_raw(const float* __restrict__ h, const float* __restrict__ wb,
                          const float* __restrict__ mask, float* __restrict__ att) {
  long i = (long)blockIdx.x * blockDim.x + threadIdx.x;
  if (i < (long)BB * NN) {
    const float* hp = h + i * DH;
    float s = 0.f;
#pragma unroll 8
    for (int d = 0; d < DH; ++d) s += hp[d] * wb[d];
    att[i] = s + (mask[i] - 1.0f) * 1.0e10f;
  }
}

__global__ void k_softmax_num(float* __restrict__ att) {   // exp(x - rowmax), TAU = 1
  __shared__ float red[256];
  int b = blockIdx.x;
  float m = -3.0e38f;
  for (int n = threadIdx.x; n < NN; n += blockDim.x) m = fmaxf(m, att[b * NN + n]);
  red[threadIdx.x] = m; __syncthreads();
  for (int o = 128; o; o >>= 1) { if (threadIdx.x < o) red[threadIdx.x] = fmaxf(red[threadIdx.x], red[threadIdx.x + o]); __syncthreads(); }
  float mx = red[0];
  for (int n = threadIdx.x; n < NN; n += blockDim.x)
    att[b * NN + n] = expf(att[b * NN + n] - mx);
}

// y[b][i] = sum_j adj[b][i][j] * x[b][j]   (f32 adj, one wave per row)
__global__ void k_spmv(const float* __restrict__ adj, const float* __restrict__ x,
                       float* __restrict__ y) {
  int gw   = (int)((blockIdx.x * blockDim.x + threadIdx.x) >> 5);
  int lane = threadIdx.x & 31;
  if (gw >= BB * NN) return;
  int b = gw / NN, i = gw - b * NN;
  const float* arow = adj + ((long)b * NN + i) * NN;
  const float* xb   = x + (long)b * NN;
  float s = 0.f;
  for (int j = lane; j < NN; j += 32) s += arow[j] * xb[j];
#pragma unroll
  for (int o = 16; o; o >>= 1) s += __shfl_xor(s, o, 32);
  if (lane == 0) y[gw] = s;
}

__global__ void k_z(const float* __restrict__ h, const float* __restrict__ attb,
                    const float* __restrict__ d2, float* __restrict__ Z) {
  long i = (long)blockIdx.x * blockDim.x + threadIdx.x;
  if (i < (long)BB * NN * DH) {
    long bn = i / DH;
    float att = attb[bn] / (d2[bn] + EPSV);
    Z[i] = att * h[i];
  }
}

// exact top-k via full bitonic sort of 2048 (value,index) pairs in LDS
__global__ void k_topk(const float* __restrict__ attb, const float* __restrict__ d2,
                       const float* __restrict__ msum, int* __restrict__ topIdx,
                       float* __restrict__ newMask) {
  __shared__ float key[NN];
  __shared__ int   idx[NN];
  int b = blockIdx.x;
  for (int n = threadIdx.x; n < NN; n += blockDim.x) {
    float att = attb[b * NN + n] / (d2[b * NN + n] + EPSV);
    key[n] = -att;            // ascending sort of (-att, idx) == stable descending top-k
    idx[n] = n;
  }
  __syncthreads();
  for (int k = 2; k <= NN; k <<= 1) {
    for (int j = k >> 1; j > 0; j >>= 1) {
      for (int i = threadIdx.x; i < NN; i += blockDim.x) {
        int l = i ^ j;
        if (l > i) {
          bool  asc = ((i & k) == 0);
          float ki = key[i], kl = key[l];
          int   ii = idx[i], il = idx[l];
          bool  gt = (ki > kl) || (ki == kl && ii > il);
          if (gt == asc) { key[i] = kl; key[l] = ki; idx[i] = il; idx[l] = ii; }
        }
      }
      __syncthreads();
    }
  }
  float kf = ceilf(0.25f * msum[b]);
  for (int t = threadIdx.x; t < KMAX; t += blockDim.x) {
    topIdx[b * KMAX + t]  = idx[t];
    newMask[b * KMAX + t] = ((float)t < kf) ? 1.0f : 0.0f;
  }
}

__global__ void k_assign_gather(const float* __restrict__ adj, const int* __restrict__ topIdx,
                                const float* __restrict__ newMask, float* __restrict__ assign) {
  int bi = blockIdx.x;                   // over B*KMAX rows
  int b = bi / KMAX;
  int src = topIdx[bi];
  float m = newMask[bi];
  const float* arow = adj + ((long)b * NN + src) * NN;
  float* dst = assign + (long)bi * NN;
  for (int j = threadIdx.x; j < NN; j += blockDim.x) dst[j] = arow[j] * m;
}

__global__ void k_colsum(const float* __restrict__ assign, float* __restrict__ cs) {
  long i = (long)blockIdx.x * blockDim.x + threadIdx.x;   // over B*NN columns
  if (i >= (long)BB * NN) return;
  long b = i / NN, j = i - b * NN;
  const float* p = assign + b * (long)KMAX * NN + j;
  float s = 0.f;
  for (int r = 0; r < KMAX; ++r) s += p[(long)r * NN];
  cs[i] = s;
}

__global__ void k_assign_norm_bf16(const float* __restrict__ assign, const float* __restrict__ cs,
                                   __bf16* __restrict__ abf) {
  long i = (long)blockIdx.x * blockDim.x + threadIdx.x;   // B*KMAX*NN
  if (i < (long)BB * KMAX * NN) {
    long b = i / ((long)KMAX * NN);
    long j = i % NN;
    abf[i] = f2bf(assign[i] / (cs[b * NN + j] + EPSV));
  }
}

// ---------------- launcher ----------------
static inline int blks(long n, int t) { return (int)((n + t - 1) / t); }

extern "C" void kernel_launch(void* const* d_in, const int* in_sizes, int n_in,
                              void* d_out, int out_size, void* d_ws, size_t ws_size,
                              hipStream_t stream) {
  const float* X    = (const float*)d_in[0];
  const float* adj  = (const float*)d_in[1];
  const float* mask = (const float*)d_in[2];
  const float* W1   = (const float*)d_in[3];
  const float* b1   = (const float*)d_in[4];
  const float* W2   = (const float*)d_in[5];
  const float* b2   = (const float*)d_in[6];
  /* w_a = d_in[7] unused by reference */
  const float* wb   = (const float*)d_in[8];

  float* out_pool = (float*)d_out;                              // (8,64)
  float* out_H    = out_pool + BB * DH;                         // (8,512,64)
  float* out_adj  = out_H + (long)BB * KMAX * DH;               // (8,512,512)
  float* out_mask = out_adj + (long)BB * KMAX * KMAX;           // (8,512)

  // workspace carve (~217 MB, 256B aligned)
  char* ws = (char*)d_ws; size_t off = 0;
  auto alloc = [&](size_t bytes) -> void* {
    void* p = ws + off; off = (off + bytes + 255) & ~(size_t)255; return p;
  };
  __bf16* adjbf  = (__bf16*)alloc((size_t)BB * NN * NN * 2);    // adj   (A operand)
  __bf16* adjTbf = (__bf16*)alloc((size_t)BB * NN * NN * 2);    // adj^T (B operand)
  __bf16* Xbf    = (__bf16*)alloc((size_t)BB * NN * DH * 2);
  __bf16* W1t    = (__bf16*)alloc((size_t)DH * DH * 2);
  __bf16* W2t    = (__bf16*)alloc((size_t)DH * DH * 2);
  float*  Tf     = (float*) alloc((size_t)BB * NN * DH * 4);    // T1 -> T2 -> Z
  __bf16* Tt     = (__bf16*)alloc((size_t)BB * DH * NN * 2);    // T1^T -> T2^T -> Z^T
  float*  Hf     = (float*) alloc((size_t)BB * NN * DH * 4);    // h1 -> h (masked)
  __bf16* Hbf    = (__bf16*)alloc((size_t)BB * NN * DH * 2);    // h1 bf16 (A operand)
  float*  attb   = (float*) alloc((size_t)BB * NN * 4);
  float*  dd1    = (float*) alloc((size_t)BB * NN * 4);
  float*  dd2    = (float*) alloc((size_t)BB * NN * 4);
  float*  msum   = (float*) alloc(256);
  int*    tidx   = (int*)   alloc((size_t)BB * KMAX * 4);
  float*  cs     = (float*) alloc((size_t)BB * NN * 4);
  float*  P      = (float*) alloc((size_t)BB * KMAX * NN * 4);  // assign f32 -> M1 f32
  __bf16* abf    = (__bf16*)alloc((size_t)BB * KMAX * NN * 2);  // normalized assign
  __bf16* m1bf   = (__bf16*)alloc((size_t)BB * KMAX * NN * 2);
  (void)ws_size; (void)n_in; (void)in_sizes; (void)out_size;

  const long nAdj = (long)BB * NN * NN, nBND = (long)BB * NN * DH, nBN = (long)BB * NN;
  const long nAsn = (long)BB * KMAX * NN;

  // operand prep: adj bf16 + adj^T bf16 (134 MB total, L2-resident across all GEMMs)
  k_cvt_bf16<<<blks(nAdj, 256), 256, 0, stream>>>(adj, adjbf, nAdj);
  { dim3 g(NN >> 5, NN >> 5, BB);
    k_cvt_bf16_t<<<g, 256, 0, stream>>>(adj, adjTbf, NN, NN, (long)NN * NN, (long)NN * NN); }
  k_cvt_bf16<<<blks(nBND, 256), 256, 0, stream>>>(X, Xbf, nBND);
  { dim3 g(2, 2, 1); k_cvt_bf16_t<<<g, 256, 0, stream>>>(W1, W1t, DH, DH, 0, 0); }
  { dim3 g(2, 2, 1); k_cvt_bf16_t<<<g, 256, 0, stream>>>(W2, W2t, DH, DH, 0, 0); }

  auto gemm = [&](const __bf16* A, const __bf16* Bt, float* Cp, const float* bias,
                  int M, int Nc, int K, int lda, int ldbt, int ldc,
                  long sA, long sB, long sC, int nb, int tnh) {
    dim3 g(Nc >> 6, M >> 6, nb);
    k_wmma_gemm<<<g, 128, 0, stream>>>(A, Bt, Cp, bias, K, lda, ldbt, ldc, sA, sB, sC, tnh);
  };
  auto cvtT = [&](const float* src, __bf16* dst) {   // (B,N,64) -> (B,64,N) bf16
    dim3 g(DH >> 5, NN >> 5, BB);
    k_cvt_bf16_t<<<g, 256, 0, stream>>>(src, dst, NN, DH, (long)NN * DH, (long)DH * NN);
  };

  // layer 1: T1 = X @ W1 ; h1 = adj @ T1 + b1
  gemm(Xbf, W1t, Tf, nullptr, BB * NN, DH, DH, DH, DH, DH, 0, 0, 0, 1, 0);
  cvtT(Tf, Tt);
  gemm(adjbf, Tt, Hf, b1, NN, DH, NN, NN, NN, DH,
       (long)NN * NN, (long)DH * NN, (long)NN * DH, BB, 0);
  // layer 2: T2 = h1 @ W2 ; h = adj @ T2 + b2
  k_cvt_bf16<<<blks(nBND, 256), 256, 0, stream>>>(Hf, Hbf, nBND);
  gemm(Hbf, W2t, Tf, nullptr, BB * NN, DH, DH, DH, DH, DH, 0, 0, 0, 1, 0);
  cvtT(Tf, Tt);
  gemm(adjbf, Tt, Hf, b2, NN, DH, NN, NN, NN, DH,
       (long)NN * NN, (long)DH * NN, (long)NN * DH, BB, 0);

  // mask, pooled mean, attention numerator, 2-hop denominator
  k_apply_mask<<<blks(nBND, 256), 256, 0, stream>>>(Hf, mask);
  k_masksum<<<BB, 256, 0, stream>>>(mask, msum);
  k_colmean<<<BB, DH, 0, stream>>>(Hf, msum, out_pool);
  k_att_raw<<<blks(nBN, 256), 256, 0, stream>>>(Hf, wb, mask, attb);
  k_softmax_num<<<BB, 256, 0, stream>>>(attb);
  k_spmv<<<blks(nBN * 32, 256), 256, 0, stream>>>(adj, attb, dd1);
  k_spmv<<<blks(nBN * 32, 256), 256, 0, stream>>>(adj, dd1, dd2);

  // Z = att * h (into Tf), transposed bf16 for the final GEMM
  k_z<<<blks(nBND, 256), 256, 0, stream>>>(Hf, attb, dd2, Tf);
  cvtT(Tf, Tt);                                                  // Z^T

  // exact top-512 + new_mask
  k_topk<<<BB, 1024, 0, stream>>>(attb, dd2, msum, tidx, out_mask);

  // assign: gather rows of adj, column-normalize, bf16
  k_assign_gather<<<BB * KMAX, 256, 0, stream>>>(adj, tidx, out_mask, P);
  k_colsum<<<blks(nBN, 256), 256, 0, stream>>>(P, cs);
  k_assign_norm_bf16<<<blks(nAsn, 256), 256, 0, stream>>>(P, cs, abf);

  // new_adj = tanh((assign @ adj) @ assign^T)
  gemm(abf, adjTbf, P, nullptr, KMAX, NN, NN, NN, NN, NN,
       (long)KMAX * NN, (long)NN * NN, (long)KMAX * NN, BB, 0);  // M1 = assign @ adj
  k_cvt_bf16<<<blks(nAsn, 256), 256, 0, stream>>>(P, m1bf, nAsn);
  gemm(m1bf, abf, out_adj, nullptr, KMAX, KMAX, NN, NN, NN, KMAX,
       (long)KMAX * NN, (long)KMAX * NN, (long)KMAX * KMAX, BB, 1);  // @ assign^T, tanh

  // H = assign @ Z
  gemm(abf, Tt, out_H, nullptr, KMAX, DH, NN, NN, NN, DH,
       (long)KMAX * NN, (long)DH * NN, (long)KMAX * DH, BB, 0);
}
